// MultiHeadAttention_1829656068264
// MI455X (gfx1250) — matrix operand order repaired and checked by reading, not compiled, not run on hardware
//
#include <hip/hip_runtime.h>
#include <hip/hip_bf16.h>

#define SEQ 2048
#define HID 1024
#define HEADS 16
#define HDIM 64

typedef __attribute__((ext_vector_type(16))) _Float16 v16h;
typedef __attribute__((ext_vector_type(8)))  _Float16 v8h;
typedef __attribute__((ext_vector_type(8)))  float    v8f;
typedef __attribute__((ext_vector_type(4)))  float    v4f;

// ---------------------------------------------------------------------------
// WMMA helper: D = A(16x32 f16) * B(32x16 f16) + C(16x16 f32)
// ---------------------------------------------------------------------------
__device__ __forceinline__ v8f wmma_f16(v16h a, v16h b, v8f c) {
  return __builtin_amdgcn_wmma_f32_16x16x32_f16(
      /*neg_a=*/false, a, /*neg_b=*/false, b,
      /*c_mod=*/(short)0, c, /*reuse_a=*/false, /*reuse_b=*/false);
}

__device__ __forceinline__ v16h cat8(v8h lo, v8h hi) {
  return __builtin_shufflevector(lo, hi, 0, 1, 2, 3, 4, 5, 6, 7,
                                         8, 9, 10, 11, 12, 13, 14, 15);
}

// ---------------------------------------------------------------------------
// f16 fragment loaders (CDNA5 ISA 7.12.2 layouts, wave32). All 16B aligned ->
// global_load_b128 / ds_load_b128, no conversions in the hot loop.
// ---------------------------------------------------------------------------
__device__ __forceinline__ v16h load_a_frag(const _Float16* __restrict__ A,
                                            int lda, int m0, int k0) {
  const int lane = threadIdx.x & 31;
  const int half = lane >> 4;
  const int m    = lane & 15;
  const _Float16* row = A + (size_t)(m0 + m) * lda + k0 + half * 8;
  v8h lo = *(const v8h*)(row);
  v8h hi = *(const v8h*)(row + 16);
  return cat8(lo, hi);
}

// B 32x16 from X^T: B[k][n] = X[n][k] (X row-major N x K); elem i<8 -> K=half*8+i, i>=8 -> K=16+half*8+(i-8)
__device__ __forceinline__ v16h load_bT_frag(const _Float16* __restrict__ X,
                                             int ldx, int n0, int k0) {
  const int lane = threadIdx.x & 31;
  const int half = lane >> 4;
  const int n    = lane & 15;
  const _Float16* row = X + (size_t)(n0 + n) * ldx + k0 + half * 8;
  v8h lo = *(const v8h*)(row);
  v8h hi = *(const v8h*)(row + 16);
  return cat8(lo, hi);
}

// ---------------------------------------------------------------------------
// elementwise f32 -> f16 (8 elems/thread)
// ---------------------------------------------------------------------------
__global__ __launch_bounds__(256)
void cvt_f32_f16(const float* __restrict__ s, _Float16* __restrict__ d, int n) {
  const size_t i = ((size_t)blockIdx.x * 256 + threadIdx.x) * 8;
  if (i >= (size_t)n) return;
  v4f a = *(const v4f*)(s + i);
  v4f b = *(const v4f*)(s + i + 4);
  v8h o;
#pragma unroll
  for (int j = 0; j < 4; ++j) { o[j] = (_Float16)a[j]; o[4 + j] = (_Float16)b[j]; }
  *(v8h*)(d + i) = o;
}

// ---------------------------------------------------------------------------
// GEMM: C[M,N] = A[M,K] @ W[N,K]^T, f16 in / OutT out.
// block = 256 (8 waves), wave tile 32x64 (8 WMMA per 32-deep K-step).
// Block tile 128x128. TRANSPOSE=true stores C^T with contiguous v8h stores.
// ---------------------------------------------------------------------------
template <typename OutT, bool TRANSPOSE>
__global__ __launch_bounds__(256)
void wmma_gemm_f16(const _Float16* __restrict__ A, const _Float16* __restrict__ W,
                   OutT* __restrict__ C, int M, int N, int K) {
  const int wave = threadIdx.x >> 5;
  const int lane = threadIdx.x & 31;
  const int half = lane >> 4;
  const int n    = lane & 15;
  const int wm   = wave & 3;
  const int wn   = wave >> 2;
  const int m0   = blockIdx.y * 128 + wm * 32;
  const int n0   = blockIdx.x * 128 + wn * 64;

  v8f z = {};
  v8f acc[2][4] = {{z, z, z, z}, {z, z, z, z}};

  for (int k0 = 0; k0 < K; k0 += 32) {
    v16h a0 = load_a_frag(A, K, m0,      k0);
    v16h a1 = load_a_frag(A, K, m0 + 16, k0);
    v16h b[4];
#pragma unroll
    for (int j = 0; j < 4; ++j) b[j] = load_bT_frag(W, K, n0 + j * 16, k0);
#pragma unroll
    for (int j = 0; j < 4; ++j) {
      acc[0][j] = wmma_f16(a0, b[j], acc[0][j]);
      acc[1][j] = wmma_f16(a1, b[j], acc[1][j]);
    }
    if (k0 + 32 < K) {
      // pull next K-slab of weight rows toward the WGP (global_prefetch_b8)
      __builtin_prefetch(W + (size_t)(n0 + lane) * K + k0 + 32, 0, 0);
    }
  }

#pragma unroll
  for (int i = 0; i < 2; ++i)
#pragma unroll
    for (int j = 0; j < 4; ++j) {
      const int mb = m0 + i * 16;
      const int nb = n0 + j * 16;
      if constexpr (TRANSPOSE) {
        v8h st;
#pragma unroll
        for (int r = 0; r < 8; ++r) st[r] = (_Float16)acc[i][j][r];
        _Float16* dst = (_Float16*)(void*)C;
        *(v8h*)(dst + (size_t)(nb + n) * M + mb + 8 * half) = st;
      } else {
#pragma unroll
        for (int r = 0; r < 8; ++r)
          C[(size_t)(mb + r + 8 * half) * N + nb + n] = (OutT)acc[i][j][r];
      }
    }
}

// ---------------------------------------------------------------------------
// RoPE on f16 (reference variant: oo uses xs_j = x[(2j-1) mod 64], m = s+1).
// One wave per (tensor, s, h) row. In-place safe (wave-wide loads precede
// stores). Q additionally absorbs log2(e)/sqrt(D) so the attention softmax
// runs natively in base-2 (raw v_exp_f32, no per-exp multiply).
// ---------------------------------------------------------------------------
__global__ __launch_bounds__(256)
void rope_kernel(_Float16* __restrict__ q, _Float16* __restrict__ k) {
  const int gid  = blockIdx.x * 8 + (threadIdx.x >> 5);
  const int lane = threadIdx.x & 31;
  const int total = 2 * SEQ * HEADS;
  if (gid >= total) return;

  const bool isQ = (gid < SEQ * HEADS);
  _Float16* base = isQ ? q : k;
  const int sh = gid % (SEQ * HEADS);
  const int s  = sh / HEADS;
  const int h  = sh % HEADS;
  _Float16* x = base + (size_t)s * HID + h * HDIM;

  const int j = lane;
  float xe = (float)x[2 * j];
  float xo = (float)x[2 * j + 1];
  float xs = (float)x[(2 * j - 1 + HDIM) & (HDIM - 1)];

  float theta = __powf(10000.0f, -2.0f * ((float)j - 1.0f) / (float)HDIM);
  float ang   = (float)(s + 1) * theta;
  float sn, cs;
  __sincosf(ang, &sn, &cs);

  // 1/sqrt(HDIM) * log2(e) folded into Q (softmax is base-change invariant)
  const float sc = isQ ? (0.125f * 1.44269504088896f) : 1.0f;
  x[2 * j]     = (_Float16)(sc * (cs * xe - sn * xo));
  x[2 * j + 1] = (_Float16)(sc * (cs * xo + sn * xs));
}

// ---------------------------------------------------------------------------
// One 32-key attention block (transposed dataflow). MASKED only for the
// single diagonal block per query tile; full blocks skip all v_cmp/v_cndmask.
// ---------------------------------------------------------------------------
template <bool MASKED>
__device__ __forceinline__ void attn_block(
    const _Float16* __restrict__ kh, const _Float16* __restrict__ vTh,
    _Float16* __restrict__ mp, v16h bq0, v16h bq1,
    int k0, int query, int half,
    float& m_run, float& l_run, v8f oacc[4]) {
  v8f zero = {};

  // ---- S^T (32 keys x 16 queries): two 16x16 tiles, keys as A-rows
  v8f s0 = zero, s1 = zero;
  {
    v16h ak00 = load_a_frag(kh, HID, k0,      0);
    v16h ak01 = load_a_frag(kh, HID, k0,      32);
    v16h ak10 = load_a_frag(kh, HID, k0 + 16, 0);
    v16h ak11 = load_a_frag(kh, HID, k0 + 16, 32);
    s0 = wmma_f16(ak00, bq0, s0);
    s0 = wmma_f16(ak01, bq1, s0);
    s1 = wmma_f16(ak10, bq0, s1);
    s1 = wmma_f16(ak11, bq1, s1);
  }

  // ---- (mask +) in-lane max over this lane's 16 keys, combine halves once
  float x0[8], x1[8];
  float mx = -1e30f;
#pragma unroll
  for (int r = 0; r < 8; ++r) {
    if constexpr (MASKED) {
      const int key = k0 + r + 8 * half;
      x0[r] = (key      <= query) ? s0[r] : -1e30f;
      x1[r] = (key + 16 <= query) ? s1[r] : -1e30f;
    } else {
      x0[r] = s0[r];
      x1[r] = s1[r];
    }
    mx = fmaxf(mx, fmaxf(x0[r], x1[r]));
  }
  mx = fmaxf(mx, __shfl_xor(mx, 16, 32));

  const float mnew  = fmaxf(m_run, mx);
  const float alpha = __builtin_amdgcn_exp2f(m_run - mnew);  // base-2 domain
  m_run = mnew;

  float rs = 0.0f;
#pragma unroll
  for (int r = 0; r < 8; ++r) {
    x0[r] = __builtin_amdgcn_exp2f(x0[r] - mnew);
    x1[r] = __builtin_amdgcn_exp2f(x1[r] - mnew);
    rs += x0[r] + x1[r];
  }
  rs += __shfl_xor(rs, 16, 32);
  l_run = l_run * alpha + rs;

#pragma unroll
  for (int t = 0; t < 4; ++t)
#pragma unroll
    for (int r = 0; r < 8; ++r) oacc[t][r] *= alpha;

  // ---- stage P^T as pq[query][key] (16x32): 2 contiguous v8h stores/lane
  const int n = threadIdx.x & 15;
  v8h pa, pb;
#pragma unroll
  for (int r = 0; r < 8; ++r) { pa[r] = (_Float16)x0[r]; pb[r] = (_Float16)x1[r]; }
  *(v8h*)(mp + n * 32 + 8 * half)      = pa;
  *(v8h*)(mp + n * 32 + 16 + 8 * half) = pb;
  asm volatile("s_wait_dscnt 0" ::: "memory");  // wave-local LDS RAW fence

  // ---- one shared P fragment, 4 O^T tiles: O^T += V^T_tile(16x32k) @ P^T
  const v16h bp = load_bT_frag(mp, 32, 0, 0);
#pragma unroll
  for (int t = 0; t < 4; ++t) {
    v16h av = load_a_frag(vTh, SEQ, t * 16, k0);
    oacc[t] = wmma_f16(av, bp, oacc[t]);
  }
}

// ---------------------------------------------------------------------------
// Causal flash attention, one wave per 16-query tile per head.
// q0/32 full (mask-free) key blocks + exactly one masked diagonal block.
// block = 256 (8 waves), grid = (SEQ/128, HEADS).
// ---------------------------------------------------------------------------
__global__ __launch_bounds__(256)
void flash_attn_kernel(const _Float16* __restrict__ q, const _Float16* __restrict__ k,
                       const _Float16* __restrict__ vT, _Float16* __restrict__ o) {
  __shared__ _Float16 pbuf[8][16 * 32];  // [query][key] per wave

  const int wave = threadIdx.x >> 5;
  const int lane = threadIdx.x & 31;
  const int half = lane >> 4;
  const int n    = lane & 15;
  const int h    = blockIdx.y;
  const int qt   = blockIdx.x * 8 + wave;
  const int q0   = qt * 16;

  const _Float16* qh  = q  + h * HDIM;
  const _Float16* kh  = k  + h * HDIM;
  const _Float16* vTh = vT + (size_t)h * HDIM * SEQ;  // [d][s], ld = SEQ
  _Float16* mp = pbuf[wave];

  // Q fragments are loop-invariant: B[d][query] for d-chunks 0..31, 32..63
  const v16h bq0 = load_bT_frag(qh, HID, q0, 0);
  const v16h bq1 = load_bT_frag(qh, HID, q0, 32);

  float m_run = -1e30f, l_run = 0.0f;   // per-lane scalars (query = q0 + n)
  v8f zero = {};
  v8f oacc[4] = {zero, zero, zero, zero};

  const int query = q0 + n;
  const int full  = q0 / 32;  // key blocks entirely below the diagonal

  for (int kb = 0; kb < full; ++kb)
    attn_block<false>(kh, vTh, mp, bq0, bq1, kb * 32, query, half,
                      m_run, l_run, oacc);
  attn_block<true>(kh, vTh, mp, bq0, bq1, full * 32, query, half,
                   m_run, l_run, oacc);

  // ---- epilogue: normalize, store row-major (s, h*64+d); contiguous v8h
  const float inv = 1.0f / l_run;
  _Float16* orow = o + (size_t)query * HID + h * HDIM;
#pragma unroll
  for (int t = 0; t < 4; ++t) {
    v8h ov;
#pragma unroll
    for (int r = 0; r < 8; ++r) ov[r] = (_Float16)(oacc[t][r] * inv);
    *(v8h*)(orow + t * 16 + 8 * half) = ov;
  }
}

// ---------------------------------------------------------------------------
// Host-side launch
// ---------------------------------------------------------------------------
extern "C" void kernel_launch(void* const* d_in, const int* in_sizes, int n_in,
                              void* d_out, int out_size, void* d_ws, size_t ws_size,
                              hipStream_t stream) {
  const float* hs = (const float*)d_in[0];
  const float* Wq = (const float*)d_in[1];
  const float* Wk = (const float*)d_in[2];
  const float* Wv = (const float*)d_in[3];
  const float* Wo = (const float*)d_in[4];
  float* out = (float*)d_out;

  // f16 workspace layout
  _Float16* hs16 = (_Float16*)d_ws;                 // SEQ*HID
  _Float16* wq16 = hs16 + (size_t)SEQ * HID;        // HID*HID
  _Float16* wk16 = wq16 + (size_t)HID * HID;
  _Float16* wv16 = wk16 + (size_t)HID * HID;
  _Float16* wo16 = wv16 + (size_t)HID * HID;
  _Float16* q16  = wo16 + (size_t)HID * HID;        // SEQ*HID
  _Float16* k16  = q16  + (size_t)SEQ * HID;        // SEQ*HID
  _Float16* vT16 = k16  + (size_t)SEQ * HID;        // HID*SEQ (transposed)
  _Float16* at16 = vT16 + (size_t)HID * SEQ;        // SEQ*HID

  // ---- one-time f32 -> f16 conversion (then everything streams f16)
  const int nHS = SEQ * HID, nW = HID * HID;
  cvt_f32_f16<<<nHS / (8 * 256), 256, 0, stream>>>(hs, hs16, nHS);
  cvt_f32_f16<<<nW  / (8 * 256), 256, 0, stream>>>(Wq, wq16, nW);
  cvt_f32_f16<<<nW  / (8 * 256), 256, 0, stream>>>(Wk, wk16, nW);
  cvt_f32_f16<<<nW  / (8 * 256), 256, 0, stream>>>(Wv, wv16, nW);
  cvt_f32_f16<<<nW  / (8 * 256), 256, 0, stream>>>(Wo, wo16, nW);

  dim3 gblk(256);
  dim3 ggrd(HID / 128, SEQ / 128);

  // ---- QKV projections (V stored transposed for the attention inner loop)
  wmma_gemm_f16<_Float16, false><<<ggrd, gblk, 0, stream>>>(hs16, wq16, q16,  SEQ, HID, HID);
  wmma_gemm_f16<_Float16, false><<<ggrd, gblk, 0, stream>>>(hs16, wk16, k16,  SEQ, HID, HID);
  wmma_gemm_f16<_Float16, true ><<<ggrd, gblk, 0, stream>>>(hs16, wv16, vT16, SEQ, HID, HID);

  // ---- RoPE on q and k (in place), 1/sqrt(D)*log2(e) folded into q
  rope_kernel<<<(2 * SEQ * HEADS) / 8, 256, 0, stream>>>(q16, k16);

  // ---- causal flash attention (transposed dataflow, split full/diag blocks)
  flash_attn_kernel<<<dim3(SEQ / 128, HEADS), 256, 0, stream>>>(q16, k16, vT16, at16);

  // ---- output projection, f32 result to d_out
  wmma_gemm_f16<float, false><<<ggrd, gblk, 0, stream>>>(at16, wo16, out, SEQ, HID, HID);
}
